// DCNPack_30236569764331
// MI455X (gfx1250) — matrix operand order repaired and checked by reading, not compiled, not run on hardware
//
#include <hip/hip_runtime.h>
#include <cmath>

#define BATCH 4
#define CINCH 64
#define HHH 160
#define WWW 160
#define HWSZ (HHH*WWW)          // 25600
#define NTOT (BATCH*HWSZ)       // 102400
#define GRP 8
#define CGRP 8
#define KKT 9
#define KDIM 576                // CIN*9
#define KSTEPS 18               // KDIM/32
#define MOFF 216
#define MT_OFF 14               // ceil(216/16)

typedef __attribute__((ext_vector_type(16))) __bf16 v16bf;
typedef __attribute__((ext_vector_type(8)))  float  v8f;

// ---------------------------------------------------------------------------
// WMMA operand register layouts (wave32, 16-bit A/B):
//  A (16x32 MxK): lane (hf=lane>>4, m=lane&15); element e=2v+o holds
//      K = e + hf*16 (contiguous, same order as B)
//  B (32x16 KxN): lane n = lane&15; element e holds K = e + hf*16 (this code's order, not AMD's table)
//  D (16x16 f32): element v holds row m = v + hf*8, col n = lane&15
// Weights are pre-swizzled by prep kernels into the exact A layout:
//  frag[mt][ks][lane][16] -> each wave loads its operand with 2x b128.
// ---------------------------------------------------------------------------

// ---------------- Prep kernels ---------------------------------------------
__global__ void prep_feat_bf16(const float* __restrict__ feat,
                               __bf16* __restrict__ featb, int n) {
  int i = (blockIdx.x * blockDim.x + threadIdx.x) * 4;
  if (i + 3 < n) {
    float4 f = *(const float4*)(feat + i);
    featb[i + 0] = (__bf16)f.x;
    featb[i + 1] = (__bf16)f.y;
    featb[i + 2] = (__bf16)f.z;
    featb[i + 3] = (__bf16)f.w;
  } else {
    for (; i < n; ++i) featb[i] = (__bf16)feat[i];
  }
}

// Offset-conv weights, K order: k = tap*64 + cin.  M padded 216->224 w/ zeros.
__global__ void prep_w_off(const float* __restrict__ w_off,
                           __bf16* __restrict__ AoffFrag) {
  int idx = blockIdx.x * blockDim.x + threadIdx.x;
  if (idx >= MT_OFF * KSTEPS * 32 * 16) return;
  int e    = idx & 15;
  int lane = (idx >> 4) & 31;
  int ksmt = idx >> 9;
  int ks = ksmt % KSTEPS, mt = ksmt / KSTEPS;
  int hf = lane >> 4, m = lane & 15;
  int v = e >> 1, ob = e & 1;
  int kb = e + (hf ? 16 : 0);
  int k = ks * 32 + kb;
  int tap = k >> 6, cin = k & 63;
  int o = mt * 16 + m;
  float val = (o < MOFF) ? w_off[((size_t)(o * CINCH + cin)) * KKT + tap] : 0.0f;
  AoffFrag[idx] = (__bf16)val;
}

// Deform-conv weights, K order: k = (g*9+tap)*8 + c, cin = g*8+c.
__global__ void prep_w_dcn(const float* __restrict__ w_dcn,
                           __bf16* __restrict__ AdcnFrag) {
  int idx = blockIdx.x * blockDim.x + threadIdx.x;
  if (idx >= 4 * KSTEPS * 32 * 16) return;
  int e    = idx & 15;
  int lane = (idx >> 4) & 31;
  int ksmt = idx >> 9;
  int ks = ksmt % KSTEPS, mt = ksmt / KSTEPS;
  int hf = lane >> 4, m = lane & 15;
  int v = e >> 1, ob = e & 1;
  int kb = e + (hf ? 16 : 0);
  int k = ks * 32 + kb;
  int g = k / 72;                 // 72 = 9 taps * 8 ch
  int tap = (k >> 3) % KKT;
  int c = k & 7;
  int cin = g * CGRP + c;
  int o = mt * 16 + m;
  AdcnFrag[idx] = (__bf16)w_dcn[((size_t)(o * CINCH + cin)) * KKT + tap];
}

// ---------------- Kernel A: offset conv, LDS-free WMMA implicit GEMM -------
// grid (800, 2), block 256 = 8 waves.  Block: M=112 (7 tiles) x N=128.
// Each wave owns 16 pixels; B fragments are built directly in registers
// (constant tap per 32-wide K chunk -> one hoisted bounds check).
__global__ __launch_bounds__(256)
void dcn_offset_gemm(const __bf16* __restrict__ featb,
                     const __bf16* __restrict__ AoffFrag,
                     const float* __restrict__ b_off,
                     float* __restrict__ offs,
                     float* __restrict__ maskb) {
  const int tid = threadIdx.x;
  const int wave = tid >> 5, lane = tid & 31;
  const int hf = lane >> 4;
  const int mt0 = blockIdx.y * 7;

  const int p = blockIdx.x * 128 + wave * 16 + (lane & 15);
  const int b = p / HWSZ, rem = p % HWSZ;
  const int y = rem / WWW, x = rem % WWW;

  v8f acc[7];
#pragma unroll
  for (int t = 0; t < 7; ++t) acc[t] = (v8f){};

  for (int ks = 0; ks < KSTEPS; ++ks) {
    const int kk0 = ks * 32;
    const int tap = kk0 >> 6;                 // constant within chunk
    const int cin0 = (kk0 & 63) + hf * 16;    // lane's first input channel
    const int yy = y + tap / 3 - 1, xx = x + tap % 3 - 1;
    const bool ok = (yy >= 0) & (yy < HHH) & (xx >= 0) & (xx < WWW);
    const __bf16* fp =
        featb + ((size_t)(b * CINCH + cin0)) * HWSZ + yy * WWW + xx;

    v16bf bfrag;
#pragma unroll
    for (int e = 0; e < 16; ++e)
      bfrag[e] = ok ? fp[(size_t)e * HWSZ] : (__bf16)0.0f;

    if (ks + 1 < KSTEPS)
      __builtin_prefetch(AoffFrag + ((((size_t)mt0 * KSTEPS + ks + 1) * 32) << 4), 0, 1);

#pragma unroll
    for (int t = 0; t < 7; ++t) {
      const v16bf a = *(const v16bf*)(
          AoffFrag + ((((size_t)(mt0 + t) * KSTEPS + ks) * 32 + lane) << 4));
      acc[t] = __builtin_amdgcn_wmma_f32_16x16x32_bf16(
          false, a, false, bfrag, (short)0, acc[t], false, false);
    }
  }

  // epilogue: bias; split into raw offsets + sigmoid mask
#pragma unroll
  for (int t = 0; t < 7; ++t) {
#pragma unroll
    for (int v = 0; v < 8; ++v) {
      int o = (mt0 + t) * 16 + v + hf * 8;
      if (o < MOFF) {
        float val = acc[t][v] + b_off[o];
        if (o < 144) {
          offs[((size_t)(b * 144 + o)) * HWSZ + rem] = val;
        } else {
          maskb[((size_t)(b * 72 + (o - 144))) * HWSZ + rem] =
              1.0f / (1.0f + __expf(-val));
        }
      }
    }
  }
}

// ---------------- Kernel B: fused deformable sampling + WMMA GEMM ----------
// grid 800, block 256.  Block: M=64 (4 tiles) x N=128.  LDS-free: each lane
// computes bilinear coords for its 2 (g,tap) combos and builds its B fragment
// in registers (8 modulated samples per combo, weights/validity hoisted).
__global__ __launch_bounds__(256)
void dcn_main_gemm(const float* __restrict__ xin,
                   const __bf16* __restrict__ AdcnFrag,
                   const float* __restrict__ offs,
                   const float* __restrict__ maskb,
                   const float* __restrict__ b_dcn,
                   float* __restrict__ out) {
  const int tid = threadIdx.x;
  const int wave = tid >> 5, lane = tid & 31;
  const int hf = lane >> 4;

  const int p = blockIdx.x * 128 + wave * 16 + (lane & 15);
  const int b = p / HWSZ, rem = p % HWSZ;
  const int y = rem / WWW, x = rem % WWW;

  v8f acc[4];
#pragma unroll
  for (int t = 0; t < 4; ++t) acc[t] = (v8f){};

  for (int ks = 0; ks < KSTEPS; ++ks) {
    const int gt0 = ks * 4;  // four (g,tap) combos per 32-wide K chunk

    v16bf bfrag;
#pragma unroll
    for (int jj = 0; jj < 2; ++jj) {
      const int gt = gt0 + hf * 2 + jj;  // lane's (g,tap) combo
      const int g = gt / KKT;
      const int tap = gt % KKT;
      const float dy = offs[((size_t)(b * 144 + gt * 2 + 0)) * HWSZ + rem];
      const float dx = offs[((size_t)(b * 144 + gt * 2 + 1)) * HWSZ + rem];
      const float mm = maskb[((size_t)(b * 72 + gt)) * HWSZ + rem];
      const float py = (float)(y + tap / 3 - 1) + dy;
      const float px = (float)(x + tap % 3 - 1) + dx;
      const float fy = floorf(py), fx = floorf(px);
      const int y0 = (int)fy, x0 = (int)fx;
      const float ly = py - fy, lx = px - fx;
      const bool vy0 = (y0 >= 0) & (y0 < HHH);
      const bool vy1 = (y0 + 1 >= 0) & (y0 + 1 < HHH);
      const bool vx0 = (x0 >= 0) & (x0 < WWW);
      const bool vx1 = (x0 + 1 >= 0) & (x0 + 1 < WWW);
      const bool ok00 = vy0 & vx0, ok01 = vy0 & vx1;
      const bool ok10 = vy1 & vx0, ok11 = vy1 & vx1;
      const int i00 = y0 * WWW + x0;
      const float w00 = (1.0f - ly) * (1.0f - lx) * mm;
      const float w01 = (1.0f - ly) * lx * mm;
      const float w10 = ly * (1.0f - lx) * mm;
      const float w11 = ly * lx * mm;
      const float* plane = xin + ((size_t)(b * CINCH + g * CGRP)) * HWSZ;
#pragma unroll
      for (int c = 0; c < 8; ++c) {
        const float* xc = plane + (size_t)c * HWSZ;
        float v00 = ok00 ? xc[i00] : 0.0f;
        float v01 = ok01 ? xc[i00 + 1] : 0.0f;
        float v10 = ok10 ? xc[i00 + WWW] : 0.0f;
        float v11 = ok11 ? xc[i00 + WWW + 1] : 0.0f;
        bfrag[jj * 8 + c] =
            (__bf16)(w00 * v00 + w01 * v01 + w10 * v10 + w11 * v11);
      }
    }

    if (ks + 1 < KSTEPS)
      __builtin_prefetch(AdcnFrag + ((((size_t)(ks + 1)) * 32) << 4), 0, 1);

#pragma unroll
    for (int t = 0; t < 4; ++t) {
      const v16bf a = *(const v16bf*)(
          AdcnFrag + ((((size_t)t * KSTEPS + ks) * 32 + lane) << 4));
      acc[t] = __builtin_amdgcn_wmma_f32_16x16x32_bf16(
          false, a, false, bfrag, (short)0, acc[t], false, false);
    }
  }

  // epilogue: bias + fp32 NCHW store
  float* op = out + (size_t)b * CINCH * HWSZ + rem;
#pragma unroll
  for (int t = 0; t < 4; ++t) {
#pragma unroll
    for (int v = 0; v < 8; ++v) {
      int m = v + hf * 8;
      op[(size_t)(t * 16 + m) * HWSZ] = acc[t][v] + b_dcn[t * 16 + m];
    }
  }
}

// ---------------- Host-side launch -----------------------------------------
extern "C" void kernel_launch(void* const* d_in, const int* in_sizes, int n_in,
                              void* d_out, int out_size, void* d_ws, size_t ws_size,
                              hipStream_t stream) {
  (void)in_sizes; (void)n_in; (void)out_size; (void)ws_size;
  const float* x     = (const float*)d_in[0];
  const float* feat  = (const float*)d_in[1];
  const float* w_off = (const float*)d_in[2];
  const float* b_off = (const float*)d_in[3];
  const float* w_dcn = (const float*)d_in[4];
  const float* b_dcn = (const float*)d_in[5];
  float* out = (float*)d_out;

  char* ws = (char*)d_ws;
  auto align256 = [](size_t v) { return (v + 255) & ~(size_t)255; };
  const size_t featN = (size_t)BATCH * CINCH * HWSZ;  // 6,553,600
  size_t o0 = 0;
  __bf16* featb = (__bf16*)(ws + o0);
  size_t o1 = align256(o0 + featN * 2);
  __bf16* AoffFrag = (__bf16*)(ws + o1);
  size_t o2 = align256(o1 + (size_t)MT_OFF * KSTEPS * 32 * 16 * 2);
  __bf16* AdcnFrag = (__bf16*)(ws + o2);
  size_t o3 = align256(o2 + (size_t)4 * KSTEPS * 32 * 16 * 2);
  float* offs = (float*)(ws + o3);
  size_t o4 = align256(o3 + (size_t)BATCH * 144 * HWSZ * 4);
  float* maskb = (float*)(ws + o4);

  prep_feat_bf16<<<(int)((featN / 4 + 255) / 256), 256, 0, stream>>>(
      feat, featb, (int)featN);
  prep_w_off<<<(MT_OFF * KSTEPS * 32 * 16 + 255) / 256, 256, 0, stream>>>(
      w_off, AoffFrag);
  prep_w_dcn<<<(4 * KSTEPS * 32 * 16 + 255) / 256, 256, 0, stream>>>(
      w_dcn, AdcnFrag);

  dim3 gA(NTOT / 128, 2);
  dcn_offset_gemm<<<gA, 256, 0, stream>>>(featb, AoffFrag, b_off, offs, maskb);

  dcn_main_gemm<<<NTOT / 128, 256, 0, stream>>>(x, AdcnFrag, offs, maskb,
                                                b_dcn, out);
}
